// SE3Transformer_16698832847174
// MI455X (gfx1250) — compile-verified, matrix operand order checked by reading, not run on hardware
//
#include <hip/hip_runtime.h>
#include <math.h>

// ---------------------------------------------------------------------------
// SE(3)-Transformer forward for MI455X (gfx1250, wave32, WMMA).
//
// Key design points (see analysis):
//  * Radial MLP 2nd layer has K=32 == native K of v_wmma_f32_16x16x32_f16:
//    one WMMA per 16-edge x 16-col tile, f32 accumulate.
//  * One wave owns 16 edges (A-fragment rows). Weights pre-converted to f16,
//    transposed [outcol][32], zero-padded to 16-col multiples; streamed from
//    L2 as B-fragments.
//  * WMMA results staged via per-wave LDS (4 waves x 16 x 192 f32 = 48KB),
//    then the equivariant contraction runs per-edge with the two lane-halves
//    splitting the 4 attention heads -> logits/values need no reductions.
//  * Segmented softmax over edge->dst with CAS float atomic-max + atomicAdd
//    (node buffers are L2-resident; 192MB L2 swallows everything node-side).
// ---------------------------------------------------------------------------

typedef _Float16 v16h __attribute__((ext_vector_type(16)));
typedef float    v8f  __attribute__((ext_vector_type(8)));

#define STAGE_COLS 192
#define ATT_WAVES  4

struct RadMLP {
    const float*    w1;   // [5][32]
    const float*    b1;   // [32]
    const _Float16* w2h;  // [pad16][32]  (transposed + f16 + zero padded)
    const float*    b2;   // [outdim]
    int outdim, pad16, ci, nj, no, di;
};

// -------------------------- per-edge geometry ------------------------------
// geom record (40 f32): [0..4]=wfeat(edge_w,r), [5]=Y0, [6..8]=Y1,
//                       [9..35]=B11[o][m][j] flat (o*3+m)*3+j
__global__ void geom_kernel(const float* __restrict__ edge_d,
                            const float* __restrict__ edge_w,
                            float* __restrict__ geom, int E)
{
    int e = blockIdx.x * 256 + threadIdx.x;
    if (e >= E) return;
    float dx = edge_d[e*3+0], dy = edge_d[e*3+1], dz = edge_d[e*3+2];
    float r  = sqrtf(dx*dx + dy*dy + dz*dz);
    float iv = 1.f / fmaxf(r, 1e-8f);
    float x = dx*iv, y = dy*iv, z = dz*iv;
    float* g = geom + (size_t)e * 40;
    g[0]=edge_w[e*4+0]; g[1]=edge_w[e*4+1]; g[2]=edge_w[e*4+2]; g[3]=edge_w[e*4+3];
    g[4]=r;
    const float Y0 = 0.28209479177387814f;
    float Y1[3] = {0.4886025119029199f*y, 0.4886025119029199f*z, 0.4886025119029199f*x};
    float Y2[5] = {1.0925484305920792f*x*y, 1.0925484305920792f*y*z,
                   0.31539156525252005f*(3.f*z*z-1.f), 1.0925484305920792f*x*z,
                   0.5462742152960396f*(x*x-y*y)};
    g[5]=Y0; g[6]=Y1[0]; g[7]=Y1[1]; g[8]=Y1[2];
    const float s2 = 0.7071067811865476f;   // 1/sqrt2
    const float s3 = 0.5773502691896258f;   // 1/sqrt3
    const float s6 = 0.4082482904638631f;   // 1/sqrt6
    float B[3][3][3];
    #pragma unroll
    for (int o=0;o<3;++o)
      #pragma unroll
      for (int m=0;m<3;++m) { B[o][m][0] = (o==m) ? Y0*s3 : 0.f; B[o][m][1]=0.f; }
    // j=1: sum_m' Y1[m'] * eps[m',o,m] / sqrt2
    B[1][2][1] =  Y1[0]*s2;  B[2][1][1] = -Y1[0]*s2;
    B[2][0][1] =  Y1[1]*s2;  B[0][2][1] = -Y1[1]*s2;
    B[0][1][1] =  Y1[2]*s2;  B[1][0][1] = -Y1[2]*s2;
    // j=2: sum_m' Y2[m'] * Q2[m',o,m]
    B[0][0][2] = -s6*Y2[2] - s2*Y2[4];
    B[0][1][2] =  s2*Y2[1];
    B[0][2][2] =  s2*Y2[0];
    B[1][0][2] =  s2*Y2[1];
    B[1][1][2] =  2.f*s6*Y2[2];
    B[1][2][2] =  s2*Y2[3];
    B[2][0][2] =  s2*Y2[0];
    B[2][1][2] =  s2*Y2[3];
    B[2][2][2] = -s6*Y2[2] + s2*Y2[4];
    #pragma unroll
    for (int o=0;o<3;++o)
      #pragma unroll
      for (int m=0;m<3;++m)
        #pragma unroll
        for (int j=0;j<3;++j) g[9 + (o*3+m)*3 + j] = B[o][m][j];
}

// ---------------- weight conversion: f32 (32,outdim) -> f16 [n][k] ---------
__global__ void convert_w2_kernel(const float* __restrict__ w2,
                                  _Float16* __restrict__ out,
                                  int outdim, int pad16)
{
    int i = blockIdx.x * 256 + threadIdx.x;
    int tot = pad16 * 32;
    if (i >= tot) return;
    int n = i >> 5, k = i & 31;
    out[n*32 + k] = (n < outdim) ? (_Float16)w2[k*outdim + n] : (_Float16)0.f;
}

// ----------------------------- WMMA helpers --------------------------------
__device__ inline v16h rad_hidden(const RadMLP& m, const float wf[5], int lane)
{
    int kbase = (lane >> 4) << 4;
    v16h a;
    #pragma unroll
    for (int i = 0; i < 16; ++i) {
        int k = kbase + i;
        float h = m.b1[k];
        #pragma unroll
        for (int t = 0; t < 5; ++t) h += wf[t] * m.w1[t*32 + k];
        a[i] = (_Float16)fmaxf(h, 0.f);
    }
    return a;
}

__device__ inline void rad_stage_tile(const RadMLP& m, v16h a, int lane,
                                      int nGlob, int nLoc,
                                      float (*stg)[STAGE_COLS])
{
    int kbase = (lane >> 4) << 4;
    const _Float16* bp = m.w2h + nGlob*32 + kbase;
    v16h b;
    #pragma unroll
    for (int i = 0; i < 16; ++i) b[i] = bp[i];
    v8f c = {};
    c = __builtin_amdgcn_wmma_f32_16x16x32_f16(false, a, false, b,
                                               (short)0, c, false, false);
    float bias = (nGlob < m.outdim) ? m.b2[nGlob] : 0.f;
    #pragma unroll
    for (int r = 0; r < 8; ++r) {
        int er = (lane < 16) ? r : (r + 8);
        stg[er][nLoc] = c[r] + bias;
    }
}

// equivariant contraction: out[c][o] = sum_{d,j} R[c,d,j] * G[d][j][o]
__device__ inline void contract_rad(const RadMLP& m, const float (*stg)[STAGE_COLS],
    int e, int c0, int srcn,
    const float* __restrict__ h0, const float* __restrict__ h1, int fin1,
    float Y0v, const float Y1v[3], const float B11[27],
    float* acc0 /*[2]*/, float (*acc1)[3] /*[2][3]*/)
{
    for (int d = 0; d < m.ci; ++d) {
        float G00 = 0.f;
        float GJ[3][3];
        if (m.di == 0) {
            float f = h0[srcn*16 + d];
            if (m.no == 1) G00 = Y0v * f;
            else { GJ[0][0]=Y1v[0]*f; GJ[0][1]=Y1v[1]*f; GJ[0][2]=Y1v[2]*f; }
        } else {
            const float* hp = h1 + ((size_t)srcn*fin1 + d)*3;
            float f0 = hp[0], f1 = hp[1], f2 = hp[2];
            if (m.no == 1) G00 = Y1v[0]*f0 + Y1v[1]*f1 + Y1v[2]*f2;
            else {
                #pragma unroll
                for (int j = 0; j < 3; ++j)
                    #pragma unroll
                    for (int o = 0; o < 3; ++o)
                        GJ[j][o] = B11[(o*3+0)*3+j]*f0 + B11[(o*3+1)*3+j]*f1
                                 + B11[(o*3+2)*3+j]*f2;
            }
        }
        #pragma unroll
        for (int cc = 0; cc < 2; ++cc) {
            int base = ((c0+cc)*m.ci + d) * m.nj;
            if (m.no == 1) {
                acc0[cc] += stg[e][base] * G00;
            } else {
                for (int j = 0; j < m.nj; ++j) {
                    float rv = stg[e][base + j];
                    #pragma unroll
                    for (int o = 0; o < 3; ++o) acc1[cc][o] += rv * GJ[j][o];
                }
            }
        }
    }
}

// ------------------------- attention edge kernel ---------------------------
struct AttnArgs {
    const float* geom; const int* src; const int* dst;
    const float* h0; const float* h1; int fin1;
    const float* q0; const float* q1;
    float* logits; float* v0; float* v1;
    int E;
    RadMLP mk[4]; RadMLP mv[4];   // order: (di,do)=(0,0),(1,0),(0,1),(1,1)
};

__global__ __launch_bounds__(ATT_WAVES*32)
void edge_attn_kernel(AttnArgs A)
{
    __shared__ float stage[ATT_WAVES][16][STAGE_COLS];
    int lane = threadIdx.x & 31;
    int wave = threadIdx.x >> 5;
    int e    = blockIdx.x * (ATT_WAVES*16) + wave*16 + (lane & 15);
    int ec   = (e < A.E) ? e : (A.E - 1);
    bool valid = (e < A.E);

    const float* gm = A.geom + (size_t)ec * 40;
    int pn = (ec + ATT_WAVES*16 < A.E) ? ec + ATT_WAVES*16 : ec;
    __builtin_prefetch(A.geom + (size_t)pn * 40, 0, 0);   // global_prefetch_b8

    float wf[5];
    #pragma unroll
    for (int i = 0; i < 5; ++i) wf[i] = gm[i];
    float Y0v = gm[5];
    float Y1v[3] = {gm[6], gm[7], gm[8]};
    float B11[27];
    #pragma unroll
    for (int i = 0; i < 27; ++i) B11[i] = gm[9+i];

    int srcn = A.src[ec];
    int dstn = A.dst[ec];
    int c0   = (lane >> 4) * 2;       // lane-half owns 2 heads/channels

    float k0[2]    = {0,0}, v0a[2]   = {0,0};
    float k1[2][3] = {{0,0,0},{0,0,0}};
    float v1a[2][3]= {{0,0,0},{0,0,0}};

    for (int m = 0; m < 4; ++m) {
        // ---- K path ----
        __syncthreads();
        {
            const RadMLP& M = A.mk[m];
            v16h a = rad_hidden(M, wf, lane);
            int colbase = lane & 15;
            for (int t = 0; t < M.pad16/16; ++t)
                rad_stage_tile(M, a, lane, t*16 + colbase, t*16 + colbase, stage[wave]);
        }
        __syncthreads();
        contract_rad(A.mk[m], stage[wave], lane & 15, c0, srcn,
                     A.h0, A.h1, A.fin1, Y0v, Y1v, B11, k0, k1);
        // ---- V path ----
        __syncthreads();
        {
            const RadMLP& M = A.mv[m];
            v16h a = rad_hidden(M, wf, lane);
            int colbase = lane & 15;
            for (int t = 0; t < M.pad16/16; ++t)
                rad_stage_tile(M, a, lane, t*16 + colbase, t*16 + colbase, stage[wave]);
        }
        __syncthreads();
        contract_rad(A.mv[m], stage[wave], lane & 15, c0, srcn,
                     A.h0, A.h1, A.fin1, Y0v, Y1v, B11, v0a, v1a);
    }

    if (valid) {
        #pragma unroll
        for (int cc = 0; cc < 2; ++cc) {
            int h = c0 + cc;
            float lg = k0[cc] * A.q0[dstn*4 + h];
            #pragma unroll
            for (int mm = 0; mm < 3; ++mm)
                lg += k1[cc][mm] * A.q1[dstn*12 + h*3 + mm];
            A.logits[(size_t)e*4 + h] = lg * 0.5f;   // /sqrt(d_k=4)
            A.v0[(size_t)e*4 + h] = v0a[cc];
            #pragma unroll
            for (int mm = 0; mm < 3; ++mm)
                A.v1[(size_t)e*12 + h*3 + mm] = v1a[cc][mm];
        }
    }
}

// --------------------------- conv edge kernel ------------------------------
struct ConvArgs {
    const float* geom; const int* src; const int* dst;
    const float* h0; const float* h1;   // both 16 channels
    float* aggc; int E;
    RadMLP r[2];                        // r0 (di=0), r1 (di=1); outdim 512
};

__global__ __launch_bounds__(ATT_WAVES*32)
void edge_conv_kernel(ConvArgs A)
{
    __shared__ float stage[ATT_WAVES][16][STAGE_COLS];
    int lane = threadIdx.x & 31;
    int wave = threadIdx.x >> 5;
    int e    = blockIdx.x * (ATT_WAVES*16) + wave*16 + (lane & 15);
    int ec   = (e < A.E) ? e : (A.E - 1);
    bool valid = (e < A.E);

    const float* gm = A.geom + (size_t)ec * 40;
    int pn = (ec + ATT_WAVES*16 < A.E) ? ec + ATT_WAVES*16 : ec;
    __builtin_prefetch(A.geom + (size_t)pn * 40, 0, 0);

    float wf[5];
    #pragma unroll
    for (int i = 0; i < 5; ++i) wf[i] = gm[i];
    float Y0v = gm[5];
    float Y1v[3] = {gm[6], gm[7], gm[8]};
    int srcn = A.src[ec];
    int dstn = A.dst[ec];
    int el   = lane & 15;
    int half = lane >> 4;

    for (int m = 0; m < 2; ++m) {
        const RadMLP& M = A.r[m];
        float G[16];
        if (M.di == 0) {
            #pragma unroll
            for (int d = 0; d < 16; ++d) G[d] = Y0v * A.h0[srcn*16 + d];
        } else {
            #pragma unroll
            for (int d = 0; d < 16; ++d) {
                const float* hp = A.h1 + ((size_t)srcn*16 + d)*3;
                G[d] = Y1v[0]*hp[0] + Y1v[1]*hp[1] + Y1v[2]*hp[2];
            }
        }
        v16h a = rad_hidden(M, wf, lane);
        int colbase = lane & 15;
        for (int ch = 0; ch < 4; ++ch) {          // 4 chunks x 128 cols = 512
            __syncthreads();
            for (int t = 0; t < 8; ++t)
                rad_stage_tile(M, a, lane, ch*128 + t*16 + colbase,
                               t*16 + colbase, stage[wave]);
            __syncthreads();
            for (int q = half; q < 8; q += 2) {   // lane-half splits channels
                int cglob = ch*8 + q;
                float acc = 0.f;
                #pragma unroll
                for (int d = 0; d < 16; ++d) acc += stage[wave][el][q*16 + d] * G[d];
                if (valid) atomicAdd(&A.aggc[(size_t)dstn*32 + cglob], acc);
            }
        }
    }
}

// ----------------------------- node kernels --------------------------------
__global__ void embed_kernel(const float* __restrict__ l0in,
                             const float* __restrict__ w1, const float* __restrict__ b1,
                             const float* __restrict__ w2, const float* __restrict__ b2,
                             float* __restrict__ h0, int N)
{
    int n = blockIdx.x * blockDim.x + threadIdx.x;
    if (n >= N) return;
    float hh[32];
    for (int j = 0; j < 32; ++j) {
        float t = b1[j];
        for (int i = 0; i < 32; ++i) t += l0in[n*32 + i] * w1[i*32 + j];
        hh[j] = (t > 0.f) ? t : (expf(t) - 1.f);   // elu
    }
    for (int c = 0; c < 16; ++c) {
        float t = b2[c];
        for (int j = 0; j < 32; ++j) t += hh[j] * w2[j*16 + c];
        h0[n*16 + c] = t;
    }
}

__global__ void q_kernel(const float* __restrict__ h0, const float* __restrict__ h1,
                         int fin1, const float* __restrict__ q0w,
                         const float* __restrict__ q1w,
                         float* __restrict__ q0, float* __restrict__ q1, int N)
{
    int n = blockIdx.x * blockDim.x + threadIdx.x;
    if (n >= N) return;
    for (int k = 0; k < 4; ++k) {
        float t = 0.f;
        for (int c = 0; c < 16; ++c) t += h0[n*16 + c] * q0w[c*4 + k];
        q0[n*4 + k] = t;
    }
    for (int k = 0; k < 4; ++k)
        for (int m = 0; m < 3; ++m) {
            float t = 0.f;
            for (int c = 0; c < fin1; ++c)
                t += h1[((size_t)n*fin1 + c)*3 + m] * q1w[c*4 + k];
            q1[n*12 + k*3 + m] = t;
        }
}

__global__ void fill_kernel(float* p, float v, int n)
{
    int i = blockIdx.x * 256 + threadIdx.x;
    if (i < n) p[i] = v;
}

__device__ inline void atomicMaxF(float* addr, float val)
{
    unsigned* ua = (unsigned*)addr;
    unsigned old = *ua;
    while (__uint_as_float(old) < val) {
        unsigned prev = atomicCAS(ua, old, __float_as_uint(val));
        if (prev == old) break;
        old = prev;
    }
}

__global__ void smax_max_kernel(const float* __restrict__ logits,
                                const int* __restrict__ dst,
                                float* __restrict__ nmax, int E)
{
    int i = blockIdx.x * 256 + threadIdx.x;
    if (i >= E*4) return;
    int e = i >> 2, h = i & 3;
    atomicMaxF(&nmax[dst[e]*4 + h], logits[i]);
}

__global__ void smax_exp_kernel(float* __restrict__ logits,
                                const int* __restrict__ dst,
                                const float* __restrict__ nmax,
                                float* __restrict__ nden, int E)
{
    int i = blockIdx.x * 256 + threadIdx.x;
    if (i >= E*4) return;
    int e = i >> 2, h = i & 3;
    float ex = expf(logits[i] - nmax[dst[e]*4 + h]);
    logits[i] = ex;                              // reuse buffer for ex
    atomicAdd(&nden[dst[e]*4 + h], ex);
}

__global__ void agg_kernel(const float* __restrict__ ex, const int* __restrict__ dst,
                           const float* __restrict__ nden,
                           const float* __restrict__ v0, const float* __restrict__ v1,
                           float* __restrict__ agg0, float* __restrict__ agg1, int E)
{
    int e = blockIdx.x * 256 + threadIdx.x;
    if (e >= E) return;
    int dn = dst[e];
    #pragma unroll
    for (int h = 0; h < 4; ++h) {
        float a = ex[(size_t)e*4 + h] / (nden[dn*4 + h] + 1e-9f);
        atomicAdd(&agg0[dn*4 + h], a * v0[(size_t)e*4 + h]);
        #pragma unroll
        for (int m = 0; m < 3; ++m)
            atomicAdd(&agg1[dn*12 + h*3 + m], a * v1[(size_t)e*12 + h*3 + m]);
    }
}

__global__ void attn_out_kernel(const float* __restrict__ agg0, const float* __restrict__ agg1,
    const float* __restrict__ h0old, const float* __restrict__ h1old, int fin1,
    const float* __restrict__ o0w, const float* __restrict__ o1w,
    const float* __restrict__ s0w, const float* __restrict__ s1w,
    const float* __restrict__ g0, const float* __restrict__ b0,
    const float* __restrict__ g1, const float* __restrict__ b1,
    float* __restrict__ h0new, float* __restrict__ h1new, int N)
{
    int n = blockIdx.x * blockDim.x + threadIdx.x;
    if (n >= N) return;
    // degree 0: proj + skip, then gnorm
    float x0[16], nrm[16];
    for (int k = 0; k < 16; ++k) {
        float t = 0.f;
        for (int c = 0; c < 4;  ++c) t += agg0[n*4 + c]  * o0w[c*16 + k];
        for (int c = 0; c < 16; ++c) t += h0old[n*16 + c] * s0w[c*16 + k];
        x0[k] = t;
    }
    float mu = 0.f;
    for (int k = 0; k < 16; ++k) { nrm[k] = sqrtf(x0[k]*x0[k] + 1e-12f); mu += nrm[k]; }
    mu *= (1.f/16.f);
    float var = 0.f;
    for (int k = 0; k < 16; ++k) { float d = nrm[k]-mu; var += d*d; }
    var *= (1.f/16.f);
    float is = rsqrtf(var + 1e-5f);
    for (int k = 0; k < 16; ++k) {
        float ln = (nrm[k]-mu)*is*g0[k] + b0[k];
        h0new[n*16 + k] = x0[k] * (fmaxf(ln, 0.f) / nrm[k]);
    }
    // degree 1
    float x1[16][3];
    for (int k = 0; k < 16; ++k)
        for (int m = 0; m < 3; ++m) {
            float t = 0.f;
            for (int c = 0; c < 4; ++c) t += agg1[n*12 + c*3 + m] * o1w[c*16 + k];
            for (int c = 0; c < fin1; ++c)
                t += h1old[((size_t)n*fin1 + c)*3 + m] * s1w[c*16 + k];
            x1[k][m] = t;
        }
    mu = 0.f;
    for (int k = 0; k < 16; ++k) {
        float s = x1[k][0]*x1[k][0] + x1[k][1]*x1[k][1] + x1[k][2]*x1[k][2];
        nrm[k] = sqrtf(s + 1e-12f); mu += nrm[k];
    }
    mu *= (1.f/16.f);
    var = 0.f;
    for (int k = 0; k < 16; ++k) { float d = nrm[k]-mu; var += d*d; }
    var *= (1.f/16.f);
    is = rsqrtf(var + 1e-5f);
    for (int k = 0; k < 16; ++k) {
        float ln = (nrm[k]-mu)*is*g1[k] + b1[k];
        float gate = fmaxf(ln, 0.f) / nrm[k];
        for (int m = 0; m < 3; ++m) h1new[(n*16 + k)*3 + m] = x1[k][m] * gate;
    }
}

__global__ void cnt_kernel(const int* __restrict__ dst, float* __restrict__ cnt, int E)
{
    int e = blockIdx.x * 256 + threadIdx.x;
    if (e < E) atomicAdd(&cnt[dst[e]], 1.f);
}

__global__ void heads_kernel(const float* __restrict__ aggc, const float* __restrict__ cnt,
    const float* __restrict__ h0, const float* __restrict__ selfw,
    const float* __restrict__ paw1, const float* __restrict__ pab1,
    const float* __restrict__ paw2, const float* __restrict__ pab2,
    const float* __restrict__ ww1,  const float* __restrict__ wb1,
    const float* __restrict__ ww2,  const float* __restrict__ wb2,
    float* __restrict__ pa, float* __restrict__ wo, int N)
{
    int n = blockIdx.x * blockDim.x + threadIdx.x;
    if (n >= N) return;
    float g[32];
    float ic = 1.f / fmaxf(cnt[n], 1.f);
    for (int k = 0; k < 32; ++k) {
        float t = aggc[(size_t)n*32 + k] * ic;
        for (int c = 0; c < 16; ++c) t += h0[n*16 + c] * selfw[c*32 + k];
        g[k] = t;
    }
    float accP = pab2[0], accW = wb2[0];
    for (int j = 0; j < 32; ++j) {
        float tp = pab1[j], tw = wb1[j];
        for (int i = 0; i < 32; ++i) {
            tp += g[i] * paw1[i*32 + j];
            tw += g[i] * ww1[i*32 + j];
        }
        accP += fmaxf(tp, 0.f) * paw2[j];
        accW += fmaxf(tw, 0.f) * ww2[j];
    }
    pa[n] = accP; wo[n] = accW;
}

__global__ void final_kernel(const float* __restrict__ idxf,
                             const float* __restrict__ pa, const float* __restrict__ wo,
                             float* __restrict__ out, int N)
{
    __shared__ float sgp[32], swp[32];
    int tid = threadIdx.x;
    int k = tid & 31, grp = tid >> 5;        // 32 cols x 8 groups
    if (tid < 32) { sgp[tid] = 0.f; swp[tid] = 0.f; }
    __syncthreads();
    float gk = 0.f, wk = 0.f;
    for (int n = grp; n < N; n += 8) {
        float iv = idxf[(size_t)n*32 + k];
        gk += iv * pa[n];
        wk += iv * wo[n];
    }
    atomicAdd(&sgp[k], gk);
    atomicAdd(&swp[k], wk);
    __syncthreads();
    if (tid < 32) out[1 + tid] = sgp[tid];
    if (tid == 0) {
        float s = 0.f;
        for (int kk = 0; kk < 32; ++kk) s += sgp[kk] * swp[kk];
        out[0] = s / 32.f;
    }
}

// ------------------------------- host side ---------------------------------
// params (d_in[7..]) assumed flattened jax-style: dicts sorted by key.
// Top order: PA, W, att0, att1, conv, lin1, lin2, norm0, norm1.
// rad dict leaves: b1,b2,w1,w2. mlp2: b1,b2,w1,w2. lin: b,w. norm: b0,b1,g0,g1.
extern "C" void kernel_launch(void* const* d_in, const int* in_sizes, int n_in,
                              void* d_out, int out_size, void* d_ws, size_t ws_size,
                              hipStream_t stream)
{
    (void)n_in; (void)out_size; (void)ws_size;
    const float* node_l0 = (const float*)d_in[0];
    const float* node_x  = (const float*)d_in[1];
    const float* edge_d  = (const float*)d_in[2];
    const float* edge_w  = (const float*)d_in[3];
    const float* idxf    = (const float*)d_in[4];
    const int*   esrc    = (const int*)d_in[5];
    const int*   edst    = (const int*)d_in[6];
    const int N = in_sizes[0] / 32;
    const int E = in_sizes[2] / 3;

    auto P = [&](int leaf) -> const float* { return (const float*)d_in[7 + leaf]; };
    // leaf index map (see analysis): PA=0..3, W=4..7, att0=8.., att1=46..,
    // conv: r0=84..87, r1=88..91, selfw=92, lin1 b=93 w=94, lin2 b=95 w=96,
    // norm0=97..100 (b0,b1,g0,g1), norm1=101..104.

    char* wsb = (char*)d_ws;
    size_t off = 0;
    auto alloc = [&](size_t bytes) -> void* {
        void* p = wsb + off;
        off = (off + bytes + 255) & ~(size_t)255;
        return p;
    };

    float* geom = (float*)alloc((size_t)E * 40 * sizeof(float));

    auto buildRad = [&](int lb, int co, int ci, int nj, int no, int di) -> RadMLP {
        RadMLP m;
        m.b1 = P(lb + 0); m.b2 = P(lb + 1); m.w1 = P(lb + 2);
        m.outdim = co * ci * nj;
        m.pad16  = (m.outdim + 15) & ~15;
        m.ci = ci; m.nj = nj; m.no = no; m.di = di;
        _Float16* wh = (_Float16*)alloc((size_t)m.pad16 * 32 * sizeof(_Float16));
        m.w2h = wh;
        int tot = m.pad16 * 32;
        convert_w2_kernel<<<(tot + 255) / 256, 256, 0, stream>>>(P(lb + 3), wh,
                                                                 m.outdim, m.pad16);
        return m;
    };

    const int fin1L[2] = {1, 16};
    RadMLP attK[2][4], attV[2][4], convR[2];
    for (int L = 0; L < 2; ++L) {
        int AT = (L == 0) ? 8 : 46;
        int f1 = fin1L[L];
        attK[L][0] = buildRad(AT + 0,  4, 16, 1, 1, 0);   // k00
        attK[L][1] = buildRad(AT + 8,  4, f1, 1, 1, 1);   // k10
        attK[L][2] = buildRad(AT + 4,  4, 16, 1, 3, 0);   // k01
        attK[L][3] = buildRad(AT + 12, 4, f1, 3, 3, 1);   // k11
        attV[L][0] = buildRad(AT + 22, 4, 16, 1, 1, 0);   // v00
        attV[L][1] = buildRad(AT + 30, 4, f1, 1, 1, 1);   // v10
        attV[L][2] = buildRad(AT + 26, 4, 16, 1, 3, 0);   // v01
        attV[L][3] = buildRad(AT + 34, 4, f1, 3, 3, 1);   // v11
    }
    convR[0] = buildRad(84, 32, 16, 1, 1, 0);
    convR[1] = buildRad(88, 32, 16, 1, 1, 1);

    float* h0e    = (float*)alloc((size_t)N * 16 * 4);
    float* h0b[2] = {(float*)alloc((size_t)N * 16 * 4), (float*)alloc((size_t)N * 16 * 4)};
    float* h1b[2] = {(float*)alloc((size_t)N * 48 * 4), (float*)alloc((size_t)N * 48 * 4)};
    float* q0   = (float*)alloc((size_t)N * 4  * 4);
    float* q1   = (float*)alloc((size_t)N * 12 * 4);
    float* logb = (float*)alloc((size_t)E * 4  * 4);
    float* v0b  = (float*)alloc((size_t)E * 4  * 4);
    float* v1b  = (float*)alloc((size_t)E * 12 * 4);
    float* nmax = (float*)alloc((size_t)N * 4  * 4);
    float* nden = (float*)alloc((size_t)N * 4  * 4);
    float* agg0 = (float*)alloc((size_t)N * 4  * 4);
    float* agg1 = (float*)alloc((size_t)N * 12 * 4);
    float* aggc = (float*)alloc((size_t)N * 32 * 4);
    float* cntb = (float*)alloc((size_t)N * 4);
    float* pab  = (float*)alloc((size_t)N * 4);
    float* wob  = (float*)alloc((size_t)N * 4);

    geom_kernel<<<(E + 255) / 256, 256, 0, stream>>>(edge_d, edge_w, geom, E);
    embed_kernel<<<(N + 127) / 128, 128, 0, stream>>>(node_l0, P(94), P(93),
                                                      P(96), P(95), h0e, N);

    const float* curH0 = h0e;
    const float* curH1 = node_x;
    for (int L = 0; L < 2; ++L) {
        int AT = (L == 0) ? 8 : 46;
        int NB = (L == 0) ? 97 : 101;
        int f1 = fin1L[L];
        q_kernel<<<(N + 127) / 128, 128, 0, stream>>>(curH0, curH1, f1,
                                                      P(AT + 18), P(AT + 19), q0, q1, N);
        fill_kernel<<<(N*4  + 255) / 256, 256, 0, stream>>>(nmax, -3.0e38f, N*4);
        fill_kernel<<<(N*4  + 255) / 256, 256, 0, stream>>>(nden, 0.f, N*4);
        fill_kernel<<<(N*4  + 255) / 256, 256, 0, stream>>>(agg0, 0.f, N*4);
        fill_kernel<<<(N*12 + 255) / 256, 256, 0, stream>>>(agg1, 0.f, N*12);

        AttnArgs A;
        A.geom = geom; A.src = esrc; A.dst = edst;
        A.h0 = curH0; A.h1 = curH1; A.fin1 = f1;
        A.q0 = q0; A.q1 = q1;
        A.logits = logb; A.v0 = v0b; A.v1 = v1b; A.E = E;
        for (int m = 0; m < 4; ++m) { A.mk[m] = attK[L][m]; A.mv[m] = attV[L][m]; }
        edge_attn_kernel<<<(E + ATT_WAVES*16 - 1) / (ATT_WAVES*16),
                           ATT_WAVES*32, 0, stream>>>(A);

        smax_max_kernel<<<(E*4 + 255) / 256, 256, 0, stream>>>(logb, edst, nmax, E);
        smax_exp_kernel<<<(E*4 + 255) / 256, 256, 0, stream>>>(logb, edst, nmax, nden, E);
        agg_kernel<<<(E + 255) / 256, 256, 0, stream>>>(logb, edst, nden, v0b, v1b,
                                                        agg0, agg1, E);
        attn_out_kernel<<<(N + 127) / 128, 128, 0, stream>>>(agg0, agg1, curH0, curH1, f1,
            P(AT + 16), P(AT + 17), P(AT + 20), P(AT + 21),
            P(NB + 2), P(NB + 0), P(NB + 3), P(NB + 1),
            h0b[L], h1b[L], N);
        curH0 = h0b[L];
        curH1 = h1b[L];
    }

    fill_kernel<<<(N*32 + 255) / 256, 256, 0, stream>>>(aggc, 0.f, N*32);
    fill_kernel<<<(N    + 255) / 256, 256, 0, stream>>>(cntb, 0.f, N);
    cnt_kernel<<<(E + 255) / 256, 256, 0, stream>>>(edst, cntb, E);

    ConvArgs CA;
    CA.geom = geom; CA.src = esrc; CA.dst = edst;
    CA.h0 = curH0; CA.h1 = curH1; CA.aggc = aggc; CA.E = E;
    CA.r[0] = convR[0]; CA.r[1] = convR[1];
    edge_conv_kernel<<<(E + ATT_WAVES*16 - 1) / (ATT_WAVES*16),
                       ATT_WAVES*32, 0, stream>>>(CA);

    heads_kernel<<<(N + 127) / 128, 128, 0, stream>>>(aggc, cntb, curH0, P(92),
        P(2), P(0), P(3), P(1), P(6), P(4), P(7), P(5), pab, wob, N);

    final_kernel<<<1, 256, 0, stream>>>(idxf, pab, wob, (float*)d_out, N);
}
